// EfficientCrossAttentionV4_29480655519766
// MI455X (gfx1250) — matrix operand order repaired and checked by reading, not compiled, not run on hardware
//
#include <hip/hip_runtime.h>
#include <hip/hip_bf16.h>

// ---------------------------------------------------------------------------
// Problem constants (from reference): B=4, N=2048, C=384, H=8, HD=48
// ---------------------------------------------------------------------------
#define Bx 4
#define Nx 2048
#define Cx 384
#define Hx 8
#define HDx 48
#define Tx (Bx * Nx)        // 8192 rows
#define DPAD 64             // HD padded to 64 for K-dim of QK^T WMMA
#define BHx (Bx * Hx)       // 32 attention problems
#define KBLK 32             // keys per flash step
#define NKB (Nx / KBLK)     // 64 key blocks
#define QTILES (Nx / 16)    // 128 query tiles per (b,h)
#define SOFTMAX_SCALE 0.14433756729740643f  // 1/sqrt(48)

typedef __attribute__((ext_vector_type(16))) _Float16 v16h;
typedef __attribute__((ext_vector_type(8)))  _Float16 v8h;
typedef __attribute__((ext_vector_type(8)))  float    v8f;

__device__ __forceinline__ v16h cat8(v8h a, v8h b) {
  v16h r;
#pragma unroll
  for (int i = 0; i < 8; ++i) { r[i] = a[i]; r[i + 8] = b[i]; }
  return r;
}

// ---------------------------------------------------------------------------
// 1) fp32 -> fp16 cast
// ---------------------------------------------------------------------------
__global__ void cast_f32_to_f16(const float* __restrict__ in,
                                _Float16* __restrict__ out, int n) {
  int i = blockIdx.x * blockDim.x + threadIdx.x;
  if (i < n) out[i] = (_Float16)in[i];
}

// ---------------------------------------------------------------------------
// 2) WMMA GEMM: Y[M,Ncols] = X[M,K] * W[Ncols,K]^T + bias   (fp16 in, fp32 out)
//    One wave computes a 16x64 tile: the A-fragment (X row block) is loaded
//    once per k-step and reused by 4 WMMAs into 4 independent accumulators
//    (breaks the accumulator RAW chain, load:WMMA ratio ~2.5 b128/WMMA).
//    A-fragment: lane(l16,hi) holds row l16, halves K = k + 16*hi + {0..15}.
//    B-fragment (W^T): lane holds 16 contiguous
//    halves of row n of W.
// ---------------------------------------------------------------------------
__global__ void __launch_bounds__(128)
wmma_gemm_bias(const _Float16* __restrict__ X,
               const _Float16* __restrict__ W,
               const float* __restrict__ bias,
               float* __restrict__ Y,
               int M, int Ncols, int K) {
  const int wave = threadIdx.x >> 5;
  const int lane = threadIdx.x & 31;
  const int tile = blockIdx.x * (blockDim.x >> 5) + wave;
  const int tilesN = Ncols >> 6;                  // 64-wide N tiles
  if (tile >= (M >> 4) * tilesN) return;          // wave-uniform guard
  const int tm = tile / tilesN, tn = tile % tilesN;
  const int l16 = lane & 15, hi = lane >> 4;

  const _Float16* xr = X + (size_t)(tm * 16 + l16) * K;
  const _Float16* wr[4];
#pragma unroll
  for (int nt = 0; nt < 4; ++nt)
    wr[nt] = W + (size_t)(tn * 64 + nt * 16 + l16) * K;

  v8f acc[4] = {};
  for (int k = 0; k < K; k += 32) {
    const v16h a = cat8(*(const v8h*)(xr + k + 16 * hi),
                        *(const v8h*)(xr + k + 16 * hi + 8));
#pragma unroll
    for (int nt = 0; nt < 4; ++nt) {
      const v16h b = *(const v16h*)(wr[nt] + k + 16 * hi);
      acc[nt] = __builtin_amdgcn_wmma_f32_16x16x32_f16(
          false, a, false, b, (short)0, acc[nt], false, false);
    }
  }
#pragma unroll
  for (int nt = 0; nt < 4; ++nt) {
    const float bv = bias[tn * 64 + nt * 16 + l16];
#pragma unroll
    for (int g = 0; g < 8; ++g) {
      const int m = g + 8 * hi;                   // C/D layout: row g (+8 hi)
      Y[(size_t)(tm * 16 + m) * Ncols + tn * 64 + nt * 16 + l16] =
          acc[nt][g] + bv;
    }
  }
}

// ---------------------------------------------------------------------------
// 3) RoPE3D + repack to fp16 [BH, N, DPAD] (zero-padded d=48..63).
//    One thread per (t, h, sub): sub<24 -> rotation pair, sub>=24 -> pad pair.
// ---------------------------------------------------------------------------
__global__ void rope3d_pack(const float* __restrict__ P,
                            const int* __restrict__ pos,
                            _Float16* __restrict__ outH, float scale) {
  int g = blockIdx.x * blockDim.x + threadIdx.x;
  if (g >= Tx * Hx * 32) return;
  const int sub = g & 31;
  const int th = g >> 5;
  const int t = th / Hx, h = th % Hx;
  const int b = t / Nx, n = t % Nx;
  _Float16* ob = outH + ((size_t)(b * Hx + h) * Nx + n) * DPAD;
  if (sub < 24) {
    const int chunk = sub >> 3, j = sub & 7;      // 3 chunks of d=16, 8 pairs
    const int d0 = chunk * 16 + j, d1 = d0 + 8;
    const float inv = __powf(100.0f, -(float)j / 8.0f);
    const float ang = (float)pos[t * 3 + chunk] * inv;
    float cs, sn; __sincosf(ang, &sn, &cs);
    const float x1 = P[(size_t)t * Cx + h * HDx + d0];
    const float x2 = P[(size_t)t * Cx + h * HDx + d1];
    ob[d0] = (_Float16)((x1 * cs - x2 * sn) * scale);
    ob[d1] = (_Float16)((x2 * cs + x1 * sn) * scale);
  } else {
    const int d = 48 + (sub - 24) * 2;            // zero pad 48..63
    ob[d] = (_Float16)0.0f; ob[d + 1] = (_Float16)0.0f;
  }
}

// ---------------------------------------------------------------------------
// 4) Pack V into the B-fragment order this code uses (contiguous k, same as P's A):
//    Vpack[(((bh*NKB + kb)*3 + nt)*32 + lane)*16 + i]
//      = V[b, key = kb*32 + 16*(lane/16) + i, h, hd = nt*16 + lane%16]
//    so each V operand in the flash kernel is one contiguous 32B load.
// ---------------------------------------------------------------------------
__global__ void vpack_kernel(const float* __restrict__ Vp,
                             _Float16* __restrict__ Vpack) {
  int idx = blockIdx.x * blockDim.x + threadIdx.x;
  if (idx >= BHx * NKB * 3 * 32 * 16) return;
  int i = idx & 15; int q = idx >> 4;
  int lane = q & 31;  q >>= 5;
  int nt = q % 3;     q /= 3;
  int kb = q % NKB;   int bh = q / NKB;
  const int hi = lane >> 4, l16 = lane & 15;
  const int key = kb * KBLK + hi * 16 + i;
  const int hd = nt * 16 + l16;
  const int b = bh / Hx, h = bh % Hx;
  Vpack[idx] = (_Float16)Vp[(size_t)(b * Nx + key) * Cx + h * HDx + hd];
}

// ---------------------------------------------------------------------------
// 5) Flash attention: one wave owns one 16-query tile of one (b,h).
//    Per 32-key step: 4 WMMAs QK^T (d padded to 64), online softmax with
//    wave32 shuffle reductions, P routed D-layout -> A-layout through a
//    per-wave LDS slab (intra-wave s_wait_dscnt), then 3 WMMAs for P*V.
// ---------------------------------------------------------------------------
__global__ void __launch_bounds__(128)
flash_attn(const _Float16* __restrict__ Qh, const _Float16* __restrict__ Kh,
           const _Float16* __restrict__ Vpack, _Float16* __restrict__ Xh) {
  __shared__ alignas(32) _Float16 plds[4][16 * 32];
  const int wave = threadIdx.x >> 5;
  const int lane = threadIdx.x & 31;
  const int tile = blockIdx.x * 4 + wave;         // exact: 1024 blocks * 4
  const int qt = tile & (QTILES - 1);
  const int bh = tile >> 7;                       // QTILES = 128
  const int l16 = lane & 15, hi = lane >> 4;

  // Q A-fragments for the whole tile (d = 0..63, zeros in pad)
  const _Float16* qb = Qh + ((size_t)bh * Nx + qt * 16 + l16) * DPAD;
  const v16h aq0 = cat8(*(const v8h*)(qb + 0 + 16 * hi),
                        *(const v8h*)(qb + 8 + 16 * hi));
  const v16h aq1 = cat8(*(const v8h*)(qb + 32 + 16 * hi),
                        *(const v8h*)(qb + 40 + 16 * hi));

  v8f o0 = {}, o1 = {}, o2 = {};
  float mi[8], li[8];
#pragma unroll
  for (int g = 0; g < 8; ++g) { mi[g] = -1e30f; li[g] = 0.0f; }

  const _Float16* kbh = Kh + (size_t)bh * Nx * DPAD;
  const _Float16* vbh = Vpack + (size_t)bh * NKB * 3 * 512;
  _Float16* lp = plds[wave];

  for (int kb = 0; kb < NKB; ++kb) {
    const int key0 = kb * KBLK;
    // ---- S = Q K^T (two 16-key subtiles, K-dim 64 = 2x32) ----
    v8f S[2];
#pragma unroll
    for (int t2 = 0; t2 < 2; ++t2) {
      const _Float16* kr = kbh + (size_t)(key0 + t2 * 16 + l16) * DPAD;
      const v16h b0 = *(const v16h*)(kr + 16 * hi);        // d 0..31
      const v16h b1 = *(const v16h*)(kr + 32 + 16 * hi);   // d 32..63
      v8f z = {};
      v8f s = __builtin_amdgcn_wmma_f32_16x16x32_f16(false, aq0, false, b0,
                                                     (short)0, z, false, false);
      s = __builtin_amdgcn_wmma_f32_16x16x32_f16(false, aq1, false, b1,
                                                 (short)0, s, false, false);
      S[t2] = s;
    }
    if (kb + 1 < NKB)                              // stream next K tile
      __builtin_prefetch(kbh + (size_t)(key0 + KBLK) * DPAD, 0, 1);

    // ---- online softmax per row (row = g + 8*hi, cols across 16 lanes) ----
#pragma unroll
    for (int g = 0; g < 8; ++g) {
      float rm = fmaxf(S[0][g], S[1][g]);
      rm = fmaxf(rm, __shfl_xor(rm, 1, 32));
      rm = fmaxf(rm, __shfl_xor(rm, 2, 32));
      rm = fmaxf(rm, __shfl_xor(rm, 4, 32));
      rm = fmaxf(rm, __shfl_xor(rm, 8, 32));       // masks < 16: stays in half
      const float mnew = fmaxf(mi[g], rm);
      const float alpha = __expf(mi[g] - mnew);
      const float p0 = __expf(S[0][g] - mnew);
      const float p1 = __expf(S[1][g] - mnew);
      float rs = p0 + p1;
      rs += __shfl_xor(rs, 1, 32);
      rs += __shfl_xor(rs, 2, 32);
      rs += __shfl_xor(rs, 4, 32);
      rs += __shfl_xor(rs, 8, 32);
      li[g] = li[g] * alpha + rs;
      mi[g] = mnew;
      o0[g] *= alpha; o1[g] *= alpha; o2[g] *= alpha;
      const int m = g + 8 * hi;                    // D-layout row
      lp[m * 32 + l16]      = (_Float16)p0;        // P -> LDS (f16)
      lp[m * 32 + 16 + l16] = (_Float16)p1;
    }
    __builtin_amdgcn_wave_barrier();
    asm volatile("s_wait_dscnt 0" ::: "memory");   // intra-wave LDS RAW

    // ---- reload P as a 16x32 A-fragment ----
    const v16h pa = cat8(*(const v8h*)(lp + l16 * 32 + 16 * hi),
                         *(const v8h*)(lp + l16 * 32 + 8 + 16 * hi));

    // ---- O += P * V (3 hd-subtiles of 16) ----
    const _Float16* vb = vbh + (size_t)kb * 3 * 512;
    const v16h v0 = *(const v16h*)(vb + 0 * 512 + lane * 16);
    const v16h v1 = *(const v16h*)(vb + 1 * 512 + lane * 16);
    const v16h v2 = *(const v16h*)(vb + 2 * 512 + lane * 16);
    o0 = __builtin_amdgcn_wmma_f32_16x16x32_f16(false, pa, false, v0,
                                                (short)0, o0, false, false);
    o1 = __builtin_amdgcn_wmma_f32_16x16x32_f16(false, pa, false, v1,
                                                (short)0, o1, false, false);
    o2 = __builtin_amdgcn_wmma_f32_16x16x32_f16(false, pa, false, v2,
                                                (short)0, o2, false, false);
  }

  // ---- epilogue: O / l, write fp16 rows into [T, C] for the Wo GEMM ----
  const int b = bh / Hx, h = bh % Hx;
#pragma unroll
  for (int g = 0; g < 8; ++g) {
    const float inv = 1.0f / li[g];
    const int m = g + 8 * hi;
    const size_t row = (size_t)(b * Nx + qt * 16 + m) * Cx + h * HDx;
    Xh[row + l16]      = (_Float16)(o0[g] * inv);
    Xh[row + 16 + l16] = (_Float16)(o1[g] * inv);
    Xh[row + 32 + l16] = (_Float16)(o2[g] * inv);
  }
}

// ---------------------------------------------------------------------------
// Host launcher
// ---------------------------------------------------------------------------
extern "C" void kernel_launch(void* const* d_in, const int* in_sizes, int n_in,
                              void* d_out, int out_size, void* d_ws,
                              size_t ws_size, hipStream_t stream) {
  const float* q   = (const float*)d_in[0];
  const float* k   = (const float*)d_in[1];
  const float* v   = (const float*)d_in[2];
  const float* Wq  = (const float*)d_in[3];
  const float* bq  = (const float*)d_in[4];
  const float* Wk  = (const float*)d_in[5];
  const float* bk  = (const float*)d_in[6];
  const float* Wv  = (const float*)d_in[7];
  const float* bv  = (const float*)d_in[8];
  const float* Wo  = (const float*)d_in[9];
  const float* bo  = (const float*)d_in[10];
  const int*   qpos = (const int*)d_in[11];
  const int*   kpos = (const int*)d_in[12];
  float* out = (float*)d_out;

  char* ws = (char*)d_ws;
  size_t off = 0;
  auto alloc = [&](size_t bytes) -> void* {
    void* p = ws + off;
    off = (off + bytes + 255) & ~(size_t)255;
    return p;
  };
  const size_t TC = (size_t)Tx * Cx;
  _Float16* hq  = (_Float16*)alloc(TC * 2);
  _Float16* hk  = (_Float16*)alloc(TC * 2);
  _Float16* hv  = (_Float16*)alloc(TC * 2);
  _Float16* hWq = (_Float16*)alloc((size_t)Cx * Cx * 2);
  _Float16* hWk = (_Float16*)alloc((size_t)Cx * Cx * 2);
  _Float16* hWv = (_Float16*)alloc((size_t)Cx * Cx * 2);
  _Float16* hWo = (_Float16*)alloc((size_t)Cx * Cx * 2);
  float* Qp = (float*)alloc(TC * 4);
  float* Kp = (float*)alloc(TC * 4);
  float* Vp = (float*)alloc(TC * 4);
  _Float16* Qh = (_Float16*)alloc((size_t)BHx * Nx * DPAD * 2);
  _Float16* Kh = (_Float16*)alloc((size_t)BHx * Nx * DPAD * 2);
  _Float16* Vpk = (_Float16*)alloc(TC * 2);
  _Float16* Xh = (_Float16*)alloc(TC * 2);

  // 1) casts
  const int nTC = (int)TC, nCC = Cx * Cx;
  cast_f32_to_f16<<<(nTC + 255) / 256, 256, 0, stream>>>(q, hq, nTC);
  cast_f32_to_f16<<<(nTC + 255) / 256, 256, 0, stream>>>(k, hk, nTC);
  cast_f32_to_f16<<<(nTC + 255) / 256, 256, 0, stream>>>(v, hv, nTC);
  cast_f32_to_f16<<<(nCC + 255) / 256, 256, 0, stream>>>(Wq, hWq, nCC);
  cast_f32_to_f16<<<(nCC + 255) / 256, 256, 0, stream>>>(Wk, hWk, nCC);
  cast_f32_to_f16<<<(nCC + 255) / 256, 256, 0, stream>>>(Wv, hWv, nCC);
  cast_f32_to_f16<<<(nCC + 255) / 256, 256, 0, stream>>>(Wo, hWo, nCC);

  // 2) Q/K/V projections (WMMA, 16x64 tile per wave)
  const int gemmBlocks = (Tx / 16) * (Cx / 64) / 4;  // 768 blocks, 4 waves ea.
  wmma_gemm_bias<<<gemmBlocks, 128, 0, stream>>>(hq, hWq, bq, Qp, Tx, Cx, Cx);
  wmma_gemm_bias<<<gemmBlocks, 128, 0, stream>>>(hk, hWk, bk, Kp, Tx, Cx, Cx);
  wmma_gemm_bias<<<gemmBlocks, 128, 0, stream>>>(hv, hWv, bv, Vp, Tx, Cx, Cx);

  // 3) RoPE3D + repack (scale folded into Q)
  const int ropeThreads = Tx * Hx * 32;
  rope3d_pack<<<ropeThreads / 256, 256, 0, stream>>>(Qp, qpos, Qh,
                                                     SOFTMAX_SCALE);
  rope3d_pack<<<ropeThreads / 256, 256, 0, stream>>>(Kp, kpos, Kh, 1.0f);

  // 4) V B-fragment pack
  const int vpackN = BHx * NKB * 3 * 32 * 16;
  vpack_kernel<<<vpackN / 256, 256, 0, stream>>>(Vp, Vpk);

  // 5) flash attention: 4096 waves = 1024 blocks * 4 waves
  flash_attn<<<(BHx * QTILES) / 4, 128, 0, stream>>>(Qh, Kh, Vpk, Xh);

  // 6) output projection into d_out
  wmma_gemm_bias<<<gemmBlocks, 128, 0, stream>>>(Xh, hWo, bo, out, Tx, Cx, Cx);
}
